// LengthRegulator_36593121362050
// MI455X (gfx1250) — compile-verified, matrix operand order checked by reading, not run on hardware
//
#include <hip/hip_runtime.h>
#include <hip/hip_bf16.h>
#include <stdint.h>

typedef __attribute__((ext_vector_type(16))) _Float16 v16h;
typedef __attribute__((ext_vector_type(8)))  _Float16 v8h;
typedef __attribute__((ext_vector_type(8)))  float    v8f;

union V16 { v16h v; v8h h[2]; };

// ---------------------------------------------------------------------------
// One-time weight preparation: W (K, D, F) f32 -> Wt (K, F, D) f16.
// Transposed so the per-(k,dc) LDS slab is contiguous along D per output
// channel, enabling async b128 copies straight into the transposed LDS tile.
// ---------------------------------------------------------------------------
__global__ void __launch_bounds__(256)
convert_transpose_w_kernel(const float* __restrict__ W, _Float16* __restrict__ Wt)
{
    constexpr int D = 256, F = 256;
    const int idx = blockIdx.x * 256 + threadIdx.x;   // over K*F*D, (k,f,d) order
    const int k = idx / (F * D);
    const int r = idx % (F * D);
    const int f = r / D;
    const int d = r % D;
    Wt[idx] = (_Float16)W[((size_t)k * D + d) * F + f];
}

// ---------------------------------------------------------------------------
// Fused conv1d(K=3, SAME) + bias + ReLU + LayerNorm over channels.
// Block: 256 threads (8 waves). Each block computes 32(T) x 256(F) of one
// batch; each wave owns a 32x32 slab as four 16x16 f32 WMMA accumulators,
// contracting K*D = 768 in chunks of 32. Weight slabs arrive via
// global_load_async_to_lds_b128 (ASYNCcnt); fragments load as ds b128 pairs.
// ---------------------------------------------------------------------------
__global__ void __launch_bounds__(256)
conv_relu_ln_kernel(const float* __restrict__ in,      // (B, T, D) f32
                    const _Float16* __restrict__ Wt,   // (K, F, D) f16
                    const float* __restrict__ bias,    // (F)
                    const float* __restrict__ gamma,   // (F)
                    const float* __restrict__ beta,    // (F)
                    float* __restrict__ out)           // (B, T, F)
{
    constexpr int T = 1024, D = 256, F = 256, K = 3;
    constexpr int AS = 264;  // a-tile row stride (halfs): 528B = 4 banks mod 64
    constexpr int WS = 56;   // w-tile row stride (halfs): 112B = 28 banks mod 64

    __shared__ _Float16 aLDS[34][AS];   // input rows t0-1 .. t0+32 (f16)
    __shared__ _Float16 wT[256][WS];    // transposed weight slab: [f][d-chunk]
    __shared__ float    outLDS[32][F];  // pre-LN tile
    __shared__ float    psum[256];
    __shared__ float    psq[256];

    const int tid  = threadIdx.x;
    const int b    = blockIdx.y;
    const int t0   = blockIdx.x * 32;
    const int wave = tid >> 5;
    const int lane = tid & 31;
    const int nrow = lane & 15;        // M row (A) / N col (B) within 16-tile
    const int hi   = lane >> 4;        // lane group: selects K sub-run
    const int f0   = wave * 32;

    // Stage 34 input rows (SAME zero padding at sequence edges) as f16.
    for (int i = tid; i < 34 * D; i += 256) {
        const int r = i >> 8, c = i & (D - 1);
        const int tg = t0 + r - 1;
        float v = 0.0f;
        if (tg >= 0 && tg < T) v = in[((size_t)b * T + tg) * D + c];
        aLDS[r][c] = (_Float16)v;
    }

    const uint32_t wt_lds_base = (uint32_t)(uintptr_t)&wT[0][0];

    v8f acc00 = {}, acc01 = {}, acc10 = {}, acc11 = {};

    for (int k = 0; k < K; ++k) {
        for (int dc = 0; dc < D; dc += 32) {
            __syncthreads();   // previous slab fully consumed
            // Async-stage 256(F) x 32(D) f16 slab: 16 KB as 1024 b128 chunks.
            // chunk c: f = c>>2, d-offset j0 = (c&3)*8.
#pragma unroll
            for (int rnd = 0; rnd < 4; ++rnd) {
                const int c  = tid + rnd * 256;
                const int f  = c >> 2;
                const int j0 = (c & 3) * 8;
                const _Float16* src = Wt + ((size_t)(k * F + f) * D + dc + j0);
                const uint32_t  dst = wt_lds_base + (uint32_t)(f * WS + j0) * 2u;
                asm volatile("global_load_async_to_lds_b128 %0, %1, off"
                             :: "v"(dst), "v"(src) : "memory");
            }
            asm volatile("s_wait_asynccnt 0x0" ::: "memory");
            __syncthreads();   // slab visible to all waves

            // A fragments (rows t0..t0+15 and t0+16..t0+31): per lane the
            // 16-bit layout is two contiguous 8-half runs -> 2 x ds b128.
            V16 a0, a1, bf0, bf1;
            {
                const int ar = nrow + k;           // aLDS row (t0-1 offset +1)
                const int c0 = dc + 8 * hi;        // K run for VGPRs 0..3
                const int c1 = dc + 16 + 8 * hi;   // K run for VGPRs 4..7
                a0.h[0] = *(const v8h*)&aLDS[ar][c0];
                a0.h[1] = *(const v8h*)&aLDS[ar][c1];
                a1.h[0] = *(const v8h*)&aLDS[ar + 16][c0];
                a1.h[1] = *(const v8h*)&aLDS[ar + 16][c1];
                // B fragments: column f0+nrow (and +16), K runs contiguous in wT.
                bf0.h[0] = *(const v8h*)&wT[f0 + nrow][8 * hi];
                bf0.h[1] = *(const v8h*)&wT[f0 + nrow][16 + 8 * hi];
                bf1.h[0] = *(const v8h*)&wT[f0 + 16 + nrow][8 * hi];
                bf1.h[1] = *(const v8h*)&wT[f0 + 16 + nrow][16 + 8 * hi];
            }

            acc00 = __builtin_amdgcn_wmma_f32_16x16x32_f16(
                false, a0.v, false, bf0.v, (short)0, acc00, false, false);
            acc01 = __builtin_amdgcn_wmma_f32_16x16x32_f16(
                false, a0.v, false, bf1.v, (short)0, acc01, false, false);
            acc10 = __builtin_amdgcn_wmma_f32_16x16x32_f16(
                false, a1.v, false, bf0.v, (short)0, acc10, false, false);
            acc11 = __builtin_amdgcn_wmma_f32_16x16x32_f16(
                false, a1.v, false, bf1.v, (short)0, acc11, false, false);
        }
    }

    // Bias + ReLU; C/D layout: VGPR r -> row r (+8 for lanes 16-31), col=lane&15.
#pragma unroll
    for (int r = 0; r < 8; ++r) {
        const int row = r + (hi ? 8 : 0);
        const int c0  = f0 + nrow;
        const int c1  = f0 + 16 + nrow;
        outLDS[row][c0]      = fmaxf(acc00[r] + bias[c0], 0.0f);
        outLDS[row][c1]      = fmaxf(acc01[r] + bias[c1], 0.0f);
        outLDS[row + 16][c0] = fmaxf(acc10[r] + bias[c0], 0.0f);
        outLDS[row + 16][c1] = fmaxf(acc11[r] + bias[c1], 0.0f);
    }
    __syncthreads();

    // LayerNorm over F=256: 8 threads per row, 32 elems per thread.
    const int rowId = tid >> 3;
    const int sub   = tid & 7;
    const int base  = sub * 32;
    float s = 0.0f, s2 = 0.0f;
#pragma unroll
    for (int c = 0; c < 32; ++c) {
        const float v = outLDS[rowId][base + c];
        s += v;
        s2 += v * v;
    }
    psum[tid] = s;
    psq[tid]  = s2;
    __syncthreads();
    for (int off = 4; off > 0; off >>= 1) {
        if (sub < off) {
            psum[tid] += psum[tid + off];
            psq[tid]  += psq[tid + off];
        }
        __syncthreads();
    }
    const float mean = psum[rowId << 3] * (1.0f / 256.0f);
    const float var  = psq[rowId << 3] * (1.0f / 256.0f) - mean * mean;
    const float rstd = rsqrtf(fmaxf(var, 0.0f) + 1e-5f);

    float* orow = out + ((size_t)b * T + t0 + rowId) * F;
#pragma unroll
    for (int c = 0; c < 32; ++c) {
        const int col = base + c;
        orow[col] = (outLDS[rowId][col] - mean) * rstd * gamma[col] + beta[col];
    }
}

// ---------------------------------------------------------------------------
// Final 256 -> 1 linear: one wave per (b,t) row, shuffle reduction.
// ---------------------------------------------------------------------------
__global__ void __launch_bounds__(256)
duration_out_kernel(const float* __restrict__ h,  // (B*T, F)
                    const float* __restrict__ Wl, // (F)
                    const float* __restrict__ bl, // (1)
                    float* __restrict__ dur)      // (B*T)
{
    constexpr int F = 256;
    const int tid  = threadIdx.x;
    const int wave = tid >> 5;
    const int lane = tid & 31;
    const long row = (long)blockIdx.x * 8 + wave;
    const float* hr = h + (size_t)row * F;
    float s = 0.0f;
#pragma unroll
    for (int i = 0; i < 8; ++i) {
        const int c = lane * 8 + i;
        s += hr[c] * Wl[c];
    }
#pragma unroll
    for (int off = 16; off > 0; off >>= 1) s += __shfl_xor(s, off, 32);
    if (lane == 0) dur[row] = s + bl[0];
}

// ---------------------------------------------------------------------------
// Inclusive scan of durations (Hillis-Steele) — one block per batch row.
// ---------------------------------------------------------------------------
__global__ void __launch_bounds__(1024)
cumsum_kernel(const int* __restrict__ target, int* __restrict__ cum)
{
    constexpr int T = 1024;
    __shared__ int s[T];
    const int b = blockIdx.x;
    const int t = threadIdx.x;
    s[t] = target[(size_t)b * T + t];
    __syncthreads();
    for (int off = 1; off < T; off <<= 1) {
        const int add = (t >= off) ? s[t - off] : 0;
        __syncthreads();
        s[t] += add;
        __syncthreads();
    }
    cum[(size_t)b * T + t] = s[t];
}

// ---------------------------------------------------------------------------
// Length regulation gather: 16 mel positions per block; binary search on the
// LDS-resident cumsum, float4-coalesced 256-float row copy with validity mask.
// ---------------------------------------------------------------------------
__global__ void __launch_bounds__(256)
gather_kernel(const float* __restrict__ x,   // (B, T, D)
              const int* __restrict__ cum,   // (B, T)
              float* __restrict__ out)       // (B, M, D)
{
    constexpr int T = 1024, D = 256, M = 4096;
    __shared__ int cumLDS[T];
    const int b   = blockIdx.y;
    const int tid = threadIdx.x;
    for (int i = tid; i < T; i += 256) cumLDS[i] = cum[(size_t)b * T + i];
    __syncthreads();

    const int total = cumLDS[T - 1];
    const int group = tid >> 4;
    const int sub   = tid & 15;
    const int m     = blockIdx.x * 16 + group;

    // idx = #(cum[t] <= m)  (upper bound on a non-decreasing array)
    int lo = 0, hi = T;
    while (lo < hi) {
        const int mid = (lo + hi) >> 1;
        if (cumLDS[mid] <= m) lo = mid + 1; else hi = mid;
    }
    const int idx = (lo < T - 1) ? lo : (T - 1);
    const float scale = (m < total) ? 1.0f : 0.0f;

    const float4* src = (const float4*)(x + ((size_t)b * T + idx) * D);
    float4*       dst = (float4*)(out + ((size_t)b * M + m) * D);
#pragma unroll
    for (int q = 0; q < 4; ++q) {
        const int e = q * 16 + sub;  // 64 contiguous floats per q across group
        float4 v = src[e];
        v.x *= scale; v.y *= scale; v.z *= scale; v.w *= scale;
        dst[e] = v;
    }
}

// ---------------------------------------------------------------------------
extern "C" void kernel_launch(void* const* d_in, const int* in_sizes, int n_in,
                              void* d_out, int out_size, void* d_ws, size_t ws_size,
                              hipStream_t stream) {
    (void)in_sizes; (void)n_in; (void)out_size; (void)ws_size;
    constexpr int B = 32, T = 1024, D = 256, F = 256, M = 4096, K = 3;

    const float* x      = (const float*)d_in[0];
    const int*   target = (const int*)d_in[1];
    // d_in[2] = mel_max_length scalar (M = 4096, fixed by problem shapes)
    const float* W1  = (const float*)d_in[3];
    const float* b1  = (const float*)d_in[4];
    const float* g1  = (const float*)d_in[5];
    const float* be1 = (const float*)d_in[6];
    const float* W2  = (const float*)d_in[7];
    const float* b2  = (const float*)d_in[8];
    const float* g2  = (const float*)d_in[9];
    const float* be2 = (const float*)d_in[10];
    const float* Wl  = (const float*)d_in[11];
    const float* bl  = (const float*)d_in[12];

    float* out = (float*)d_out;                       // (B, M, D) = 33.5M floats
    float* dur = out + (size_t)B * M * D;             // (B, T) tail of d_out
    // Activation scratch inside the not-yet-written gather output region;
    // fully overwritten by gather_kernel afterwards (stream ordered).
    float* h1 = out;                                  // (B, T, F)
    float* h2 = out + (size_t)B * T * F;

    // Workspace: cumsum (128 KB) + two f16 transposed weight tensors (384 KB ea).
    int*      cum = (int*)d_ws;
    _Float16* W1t = (_Float16*)((char*)d_ws + (size_t)B * T * sizeof(int));
    _Float16* W2t = W1t + (size_t)K * F * D;

    convert_transpose_w_kernel<<<(K * F * D) / 256, 256, 0, stream>>>(W1, W1t);
    convert_transpose_w_kernel<<<(K * F * D) / 256, 256, 0, stream>>>(W2, W2t);

    conv_relu_ln_kernel<<<dim3(T / 32, B), 256, 0, stream>>>(x,  W1t, b1, g1, be1, h1);
    conv_relu_ln_kernel<<<dim3(T / 32, B), 256, 0, stream>>>(h1, W2t, b2, g2, be2, h2);
    duration_out_kernel<<<(B * T) / 8, 256, 0, stream>>>(h2, Wl, bl, dur);
    cumsum_kernel<<<B, 1024, 0, stream>>>(target, cum);
    gather_kernel<<<dim3(M / 16, B), 256, 0, stream>>>(x, cum, out);
}
